// Decoder_12403865551638
// MI455X (gfx1250) — matrix operand order repaired and checked by reading, not compiled, not run on hardware
//
#include <hip/hip_runtime.h>
#include <hip/hip_bf16.h>

// ---------------------------------------------------------------------------
// Types for CDNA5 WMMA (wave32): D(16x16 f32) = A(16x32 bf16) * B(32x16 bf16) + C
// ---------------------------------------------------------------------------
typedef __attribute__((ext_vector_type(16))) __bf16 v16bf;
typedef __attribute__((ext_vector_type(8)))  __bf16 v8bf;
typedef __attribute__((ext_vector_type(8)))  float  v8f;
typedef __attribute__((__vector_size__(16))) int    v4i_;   // b128 payload type

#define TT 64
#define BB 64
#define EE 512
#define HH 1024
#define VV 32000
#define STEPS 63          // T-1
#define GATES 4096        // 4*H

#define GAS __attribute__((address_space(1)))
#define LAS __attribute__((address_space(3)))

// Async-to-LDS path availability (device pass only; host pass sees 0 but the
// kernel signatures/launches are identical either way).
#if defined(__AMDGCN__) && __has_builtin(__builtin_amdgcn_global_load_async_to_lds_b128)
#define ASYNC_LDS 1
#else
#define ASYNC_LDS 0
#endif

#if defined(__AMDGCN__)
#if __has_builtin(__builtin_amdgcn_s_wait_asynccnt)
#define WAIT_ASYNC0() __builtin_amdgcn_s_wait_asynccnt(0)
#else
#define WAIT_ASYNC0() asm volatile("s_wait_asynccnt 0x0" ::: "memory")
#endif
#else
#define WAIT_ASYNC0() ((void)0)
#endif

// ---------------------------------------------------------------------------
// Embedding gather + fp32->bf16 convert for all 63 timesteps
// ---------------------------------------------------------------------------
__global__ void embed_kernel(const int* __restrict__ dec,
                             const float* __restrict__ embW,
                             __bf16* __restrict__ X)
{
    int idx = blockIdx.x * blockDim.x + threadIdx.x;     // < 63*64*512
    if (idx >= STEPS * BB * EE) return;
    int e  = idx & (EE - 1);
    int tb = idx >> 9;                                   // t*64+b
    int tok = dec[tb];
    X[idx] = (__bf16)embW[(size_t)tok * EE + e];
}

// ---------------------------------------------------------------------------
// fp32 -> bf16 bulk convert (one-time weight conversion when ws allows)
// ---------------------------------------------------------------------------
__global__ void cvt_kernel(const float* __restrict__ src,
                           __bf16* __restrict__ dst, int n)
{
    int i = blockIdx.x * blockDim.x + threadIdx.x;
    if (i < n) dst[i] = (__bf16)src[i];
}

// ---------------------------------------------------------------------------
// State init: h0 (fp32) -> bf16 working state, c0 -> fp32 working state
// ---------------------------------------------------------------------------
__global__ void init_kernel(const float* __restrict__ h0,
                            const float* __restrict__ c0,
                            __bf16* __restrict__ h0bf, __bf16* __restrict__ h1bf,
                            float* __restrict__ c0f, float* __restrict__ c1f)
{
    int i = blockIdx.x * blockDim.x + threadIdx.x;       // < 64*1024
    if (i >= BB * HH) return;
    h0bf[i] = (__bf16)h0[i];
    h1bf[i] = (__bf16)h0[BB * HH + i];
    c0f[i] = c0[i];
    c1f[i] = c0[BB * HH + i];
}

// ---------------------------------------------------------------------------
// B fragment builder (32x16): lane<16 -> K {0..7,16..23}, lane>=16 -> +8
// ---------------------------------------------------------------------------
template <bool BF16W>
__device__ __forceinline__ v16bf load_bfrag(const float* Bf, const __bf16* Bh,
                                            size_t rowOff)
{
    v16bf b;
    if (BF16W) {
        const __bf16* wrow = Bh + rowOff;
        __builtin_prefetch(wrow + 32, 0, 1);
        v8bf blo = *(const v8bf*)(wrow);
        v8bf bhi = *(const v8bf*)(wrow + 16);
        b = __builtin_shufflevector(blo, bhi,
                0,1,2,3,4,5,6,7,8,9,10,11,12,13,14,15);
    } else {
        const float* wrow = Bf + rowOff;
        __builtin_prefetch(wrow + 32, 0, 1);
        v8f w0 = *(const v8f*)(wrow);
        v8f w1 = *(const v8f*)(wrow + 16);
#pragma unroll
        for (int i = 0; i < 8; ++i) {
            b[i]     = (__bf16)w0[i];
            b[8 + i] = (__bf16)w1[i];
        }
    }
    return b;
}

// ---------------------------------------------------------------------------
// Split-K WMMA GEMM for the gate matrices:
//   D(64 x N) = [A1|A2](bf16 row-major) * [B1|B2]^T, wave tile 16 x (16*NT)
// ---------------------------------------------------------------------------
template <int NT, bool BF16W>
__global__ void __launch_bounds__(256)
gemm_wmma(const __bf16* __restrict__ A1, int K1,
          const __bf16* __restrict__ A2, int K2,
          const float* __restrict__ Bf1, const float* __restrict__ Bf2,
          const __bf16* __restrict__ Bh1, const __bf16* __restrict__ Bh2,
          float* __restrict__ D, int N, int tilesN, int totalWaves)
{
    int wl = blockIdx.x * (blockDim.x >> 5) + (threadIdx.x >> 5);
    if (wl >= totalWaves) return;                        // wave-uniform guard
    int m0 = (wl / tilesN) * 16;
    int n0 = (wl % tilesN) * (16 * NT);

    int lane = threadIdx.x & 31;
    int mr   = lane & 15;
    int hi   = lane >> 4;

    v8f acc[NT];
#pragma unroll
    for (int j = 0; j < NT; ++j) acc[j] = (v8f)(0.0f);

    int K = K1 + K2;
    for (int kc = 0; kc < K; kc += 32) {
        const __bf16* Ap; const float* Bfp; const __bf16* Bhp; int ld, kk;
        if (kc < K1) { Ap = A1; Bfp = Bf1; Bhp = Bh1; ld = K1; kk = kc; }
        else         { Ap = A2; Bfp = Bf2; Bhp = Bh2; ld = K2; kk = kc - K1; }

        // A fragment (16x32): lane<16 -> K {0..7,16..23}, lane>=16 -> +8
        const __bf16* arow = Ap + (size_t)(m0 + mr) * ld + kk + hi * 8;
        v8bf alo = *(const v8bf*)(arow);
        v8bf ahi = *(const v8bf*)(arow + 16);
        v16bf a = __builtin_shufflevector(alo, ahi,
                    0,1,2,3,4,5,6,7,8,9,10,11,12,13,14,15);

        int kbB = kk + hi * 8;
#pragma unroll
        for (int j = 0; j < NT; ++j) {
            v16bf b = load_bfrag<BF16W>(Bfp, Bhp,
                          (size_t)(n0 + j * 16 + mr) * ld + kbB);
            acc[j] = __builtin_amdgcn_wmma_f32_16x16x32_bf16(
                         false, a, false, b, (short)0, acc[j], false, false);
        }
    }

#pragma unroll
    for (int j = 0; j < NT; ++j)
#pragma unroll
        for (int r = 0; r < 8; ++r)
            D[(size_t)(m0 + r + hi * 8) * N + (n0 + j * 16 + mr)] = acc[j][r];
}

// ---------------------------------------------------------------------------
// Projection GEMM: D(64 x N) = A(64 x K bf16) * B^T, B = N x K weights.
// Block = 128 threads = 4 waves; each wave owns a 64x64 tile (full M), so
// every weight element is read exactly once per step. The shared A K-chunk
// (64x32 bf16 = 4 KB) is staged into LDS with double-buffered async DMA
// (GLOBAL_LOAD_ASYNC_TO_LDS, ASYNCcnt) overlapping the WMMAs; falls back to
// direct global loads if the builtin is unavailable.
// ---------------------------------------------------------------------------
template <bool BF16W>
__global__ void __launch_bounds__(128)
gemm_proj(const __bf16* __restrict__ A, int K,
          const float* __restrict__ Bf, const __bf16* __restrict__ Bh,
          float* __restrict__ D, int N)
{
    int wave = threadIdx.x >> 5;
    int n0   = (blockIdx.x * 4 + wave) * 64;             // N % 256 == 0
    int lane = threadIdx.x & 31;
    int mr   = lane & 15;
    int hi   = lane >> 4;

    v8f acc[4][4];
#pragma unroll
    for (int m = 0; m < 4; ++m)
#pragma unroll
        for (int j = 0; j < 4; ++j) acc[m][j] = (v8f)(0.0f);

#if ASYNC_LDS
    __shared__ __align__(16) __bf16 sA[2][64 * 32];      // 2 x 4 KB
    auto issueA = [&](int kc, int buf) {
        int row = threadIdx.x >> 1;                      // 128 thr -> 64 rows x 2
        int seg = threadIdx.x & 1;
        const __bf16* g = A + (size_t)row * K + kc + seg * 16;
        __bf16* l = &sA[buf][row * 32 + seg * 16];
        GAS void* gv0 = (GAS void*)(void*)const_cast<__bf16*>(g);
        GAS void* gv1 = (GAS void*)(void*)const_cast<__bf16*>(g + 8);
        LAS void* lv0 = (LAS void*)l;
        LAS void* lv1 = (LAS void*)(l + 8);
        __builtin_amdgcn_global_load_async_to_lds_b128(
            (GAS v4i_*)gv0, (LAS v4i_*)lv0, 0, 0);
        __builtin_amdgcn_global_load_async_to_lds_b128(
            (GAS v4i_*)gv1, (LAS v4i_*)lv1, 0, 0);
    };
    issueA(0, 0);
#endif

    for (int kc = 0; kc < K; kc += 32) {
        v16bf a[4];
#if ASYNC_LDS
        int buf = (kc >> 5) & 1;
        WAIT_ASYNC0();                                   // this buffer's DMA done
        __syncthreads();                                 // all waves done w/ other buf
        if (kc + 32 < K) issueA(kc + 32, buf ^ 1);       // prefetch next chunk
#pragma unroll
        for (int m = 0; m < 4; ++m) {
            const __bf16* ap = &sA[buf][(m * 16 + mr) * 32 + hi * 8];
            v8bf alo = *(const v8bf*)(ap);
            v8bf ahi = *(const v8bf*)(ap + 16);
            a[m] = __builtin_shufflevector(alo, ahi,
                     0,1,2,3,4,5,6,7,8,9,10,11,12,13,14,15);
        }
#else
#pragma unroll
        for (int m = 0; m < 4; ++m) {
            const __bf16* arow = A + (size_t)(m * 16 + mr) * K + kc + hi * 8;
            v8bf alo = *(const v8bf*)(arow);
            v8bf ahi = *(const v8bf*)(arow + 16);
            a[m] = __builtin_shufflevector(alo, ahi,
                     0,1,2,3,4,5,6,7,8,9,10,11,12,13,14,15);
        }
#endif
        int kbB = kc + hi * 8;
#pragma unroll
        for (int j = 0; j < 4; ++j) {
            v16bf b = load_bfrag<BF16W>(Bf, Bh,
                          (size_t)(n0 + j * 16 + mr) * K + kbB);
#pragma unroll
            for (int m = 0; m < 4; ++m) {
                acc[m][j] = __builtin_amdgcn_wmma_f32_16x16x32_bf16(
                                false, a[m], false, b, (short)0, acc[m][j],
                                false, false);
            }
        }
    }

#pragma unroll
    for (int m = 0; m < 4; ++m)
#pragma unroll
        for (int j = 0; j < 4; ++j)
#pragma unroll
            for (int r = 0; r < 8; ++r)
                D[(size_t)(m * 16 + r + hi * 8) * N +
                  (n0 + j * 16 + mr)] = acc[m][j][r];
}

// ---------------------------------------------------------------------------
// LSTM cell elementwise (gate order i|f|g|o per the reference split)
// ---------------------------------------------------------------------------
__device__ __forceinline__ float sigmoidf_(float x) { return 1.0f / (1.0f + expf(-x)); }

__global__ void lstm_cell(const float* __restrict__ G,
                          const float* __restrict__ bih,
                          const float* __restrict__ bhh,
                          float* __restrict__ c,
                          __bf16* __restrict__ hbf)
{
    int idx = blockIdx.x * blockDim.x + threadIdx.x;     // < 64*1024
    if (idx >= BB * HH) return;
    int b = idx >> 10;
    int h = idx & (HH - 1);
    const float* g = G + (size_t)b * GATES;
    float gi = g[h]          + bih[h]          + bhh[h];
    float gf = g[HH + h]     + bih[HH + h]     + bhh[HH + h];
    float gg = g[2 * HH + h] + bih[2 * HH + h] + bhh[2 * HH + h];
    float go = g[3 * HH + h] + bih[3 * HH + h] + bhh[3 * HH + h];
    float cn = sigmoidf_(gf) * c[idx] + sigmoidf_(gi) * tanhf(gg);
    float hn = sigmoidf_(go) * tanhf(cn);
    c[idx]  = cn;
    hbf[idx] = (__bf16)hn;
}

// ---------------------------------------------------------------------------
// Per-row log-softmax NLL; per-(step,row) slots keep the final sum deterministic
// ---------------------------------------------------------------------------
__global__ void __launch_bounds__(256)
loss_kernel(const float* __restrict__ logits,
            const float* __restrict__ bout,
            const int* __restrict__ dec, int step,
            float* __restrict__ parts)
{
    int b = blockIdx.x;
    const float* row = logits + (size_t)b * VV;
    __shared__ float red[256];

    float m = -3.402823466e+38f;
    for (int v = threadIdx.x; v < VV; v += 256)
        m = fmaxf(m, row[v] + bout[v]);
    red[threadIdx.x] = m;
    __syncthreads();
    for (int s = 128; s > 0; s >>= 1) {
        if (threadIdx.x < s)
            red[threadIdx.x] = fmaxf(red[threadIdx.x], red[threadIdx.x + s]);
        __syncthreads();
    }
    m = red[0];
    __syncthreads();

    float sum = 0.0f;
    for (int v = threadIdx.x; v < VV; v += 256)
        sum += expf(row[v] + bout[v] - m);
    red[threadIdx.x] = sum;
    __syncthreads();
    for (int s = 128; s > 0; s >>= 1) {
        if (threadIdx.x < s)
            red[threadIdx.x] += red[threadIdx.x + s];
        __syncthreads();
    }

    if (threadIdx.x == 0) {
        int t = dec[(step + 1) * BB + b];
        float lt = row[t] + bout[t];
        parts[step * BB + b] = (logf(red[0]) + m) - lt;
    }
}

__global__ void finalize_kernel(const float* __restrict__ parts, float* __restrict__ out)
{
    if (threadIdx.x == 0 && blockIdx.x == 0) {
        float s = 0.0f;
        for (int i = 0; i < STEPS * BB; ++i) s += parts[i];
        out[0] = s;
    }
}

// ---------------------------------------------------------------------------
// Host driver
// ---------------------------------------------------------------------------
extern "C" void kernel_launch(void* const* d_in, const int* in_sizes, int n_in,
                              void* d_out, int out_size, void* d_ws, size_t ws_size,
                              hipStream_t stream)
{
    const int*   dec  = (const int*)d_in[0];
    // d_in[1] = enc_outputs: unused by the reference
    const float* h0in = (const float*)d_in[2];
    const float* c0in = (const float*)d_in[3];
    const float* embW = (const float*)d_in[4];
    const float* Wih0 = (const float*)d_in[5];
    const float* Whh0 = (const float*)d_in[6];
    const float* bih0 = (const float*)d_in[7];
    const float* bhh0 = (const float*)d_in[8];
    const float* Wih1 = (const float*)d_in[9];
    const float* Whh1 = (const float*)d_in[10];
    const float* bih1 = (const float*)d_in[11];
    const float* bhh1 = (const float*)d_in[12];
    const float* Wout = (const float*)d_in[13];
    const float* bout = (const float*)d_in[14];
    float* out = (float*)d_out;
    (void)in_sizes; (void)n_in; (void)out_size;

    char* p = (char*)d_ws;
    size_t used = 0;
    auto carve = [&](size_t bytes) -> char* {
        char* r = p;
        size_t pad = (bytes + 255) & ~(size_t)255;
        p += pad; used += pad;
        return r;
    };
    __bf16* Xbf    = (__bf16*)carve((size_t)STEPS * BB * EE * sizeof(__bf16));
    __bf16* h0bf   = (__bf16*)carve((size_t)BB * HH * sizeof(__bf16));
    __bf16* h1bf   = (__bf16*)carve((size_t)BB * HH * sizeof(__bf16));
    float*  c0f    = (float*)carve((size_t)BB * HH * sizeof(float));
    float*  c1f    = (float*)carve((size_t)BB * HH * sizeof(float));
    float*  G      = (float*)carve((size_t)BB * GATES * sizeof(float));
    float*  logits = (float*)carve((size_t)BB * VV * sizeof(float));
    float*  parts  = (float*)carve((size_t)STEPS * BB * sizeof(float));

    // Optional one-time bf16 weight copies (~95 MB): use iff workspace allows.
    const size_t nIh0 = (size_t)GATES * EE;
    const size_t nHh  = (size_t)GATES * HH;
    const size_t nOut = (size_t)VV * HH;
    const size_t wbytes = ((nIh0 + 3 * nHh + nOut) * sizeof(__bf16) + 5 * 256);
    bool bf16w = (ws_size >= used + wbytes);

    __bf16 *Wih0b = nullptr, *Whh0b = nullptr, *Wih1b = nullptr,
           *Whh1b = nullptr, *Woutb = nullptr;
    if (bf16w) {
        Wih0b = (__bf16*)carve(nIh0 * sizeof(__bf16));
        Whh0b = (__bf16*)carve(nHh  * sizeof(__bf16));
        Wih1b = (__bf16*)carve(nHh  * sizeof(__bf16));
        Whh1b = (__bf16*)carve(nHh  * sizeof(__bf16));
        Woutb = (__bf16*)carve(nOut * sizeof(__bf16));
        cvt_kernel<<<(int)((nIh0 + 255) / 256), 256, 0, stream>>>(Wih0, Wih0b, (int)nIh0);
        cvt_kernel<<<(int)((nHh  + 255) / 256), 256, 0, stream>>>(Whh0, Whh0b, (int)nHh);
        cvt_kernel<<<(int)((nHh  + 255) / 256), 256, 0, stream>>>(Wih1, Wih1b, (int)nHh);
        cvt_kernel<<<(int)((nHh  + 255) / 256), 256, 0, stream>>>(Whh1, Whh1b, (int)nHh);
        cvt_kernel<<<(int)((nOut + 255) / 256), 256, 0, stream>>>(Wout, Woutb, (int)nOut);
    }

    embed_kernel<<<(STEPS * BB * EE + 255) / 256, 256, 0, stream>>>(dec, embW, Xbf);
    init_kernel<<<(BB * HH + 255) / 256, 256, 0, stream>>>(h0in, c0in, h0bf, h1bf, c0f, c1f);

    // Gates: 16x32 wave tiles -> tilesN = 4096/32 = 128, waves = 4*128 = 512
    // Proj : 64x64 wave tiles, 4 waves/block sharing LDS A -> 125 blocks
    const int gTilesN = GATES / 32, gWaves = 4 * gTilesN, gBlocks = (gWaves + 7) / 8;
    const int pBlocks = VV / 256;

    for (int t = 0; t < STEPS; ++t) {
        const __bf16* Xt = Xbf + (size_t)t * BB * EE;
        if (bf16w) {
            gemm_wmma<2, true><<<gBlocks, 256, 0, stream>>>(
                Xt, EE, h0bf, HH, nullptr, nullptr, Wih0b, Whh0b,
                G, GATES, gTilesN, gWaves);
            lstm_cell<<<(BB * HH + 255) / 256, 256, 0, stream>>>(G, bih0, bhh0, c0f, h0bf);
            gemm_wmma<2, true><<<gBlocks, 256, 0, stream>>>(
                h0bf, HH, h1bf, HH, nullptr, nullptr, Wih1b, Whh1b,
                G, GATES, gTilesN, gWaves);
            lstm_cell<<<(BB * HH + 255) / 256, 256, 0, stream>>>(G, bih1, bhh1, c1f, h1bf);
            gemm_proj<true><<<pBlocks, 128, 0, stream>>>(
                h1bf, HH, nullptr, Woutb, logits, VV);
        } else {
            gemm_wmma<2, false><<<gBlocks, 256, 0, stream>>>(
                Xt, EE, h0bf, HH, Wih0, Whh0, nullptr, nullptr,
                G, GATES, gTilesN, gWaves);
            lstm_cell<<<(BB * HH + 255) / 256, 256, 0, stream>>>(G, bih0, bhh0, c0f, h0bf);
            gemm_wmma<2, false><<<gBlocks, 256, 0, stream>>>(
                h0bf, HH, h1bf, HH, Wih1, Whh1, nullptr, nullptr,
                G, GATES, gTilesN, gWaves);
            lstm_cell<<<(BB * HH + 255) / 256, 256, 0, stream>>>(G, bih1, bhh1, c1f, h1bf);
            gemm_proj<false><<<pBlocks, 128, 0, stream>>>(
                h1bf, HH, Wout, nullptr, logits, VV);
        }
        loss_kernel<<<BB, 256, 0, stream>>>(logits, bout, dec, t, parts);
    }

    finalize_kernel<<<1, 32, 0, stream>>>(parts, out);
}
